// CastratedGAT_52871047413955
// MI455X (gfx1250) — matrix operand order repaired and checked by reading, not compiled, not run on hardware
//
#include <hip/hip_runtime.h>
#include <hip/hip_bf16.h>

// ---------------------------------------------------------------------------
// GATConv on MI455X (gfx1250, wave32).
// N=50000 nodes, E=800000 edges, F_IN=128, H=8 heads, C=32 (hidden 256).
// GEMM h = x @ W done with v_wmma_f32_16x16x32_bf16 (fp32 accumulate).
// Everything else is L2-resident scatter/gather (h and out are 51.2MB each,
// fitting in the 192MB L2), using native global_atomic_add_f32.
// ---------------------------------------------------------------------------

#define GAT_N 50000
#define GAT_E 800000
#define GAT_FIN 128
#define GAT_H 8
#define GAT_C 32
#define GAT_F 256            // H*C
#define ROW_TILES (GAT_N / 16)   // 3125 exact
#define COL_TILES (GAT_F / 16)   // 16
#define K_CHUNKS (GAT_FIN / 32)  // 4

typedef __attribute__((ext_vector_type(16))) __bf16 v16bf;
typedef __attribute__((ext_vector_type(8)))  float  v8f;

// ---------------- workspace layout (bytes) ----------------
#define OFF_H      0ull                               // N*256 f32 = 51,200,000
#define OFF_WP     51200000ull                        // 16*4*32*16 bf16 = 65,536
#define OFF_ASRC   51265536ull                        // N*8 f32 = 1,600,000
#define OFF_ADST   52865536ull
#define OFF_M      54465536ull
#define OFF_DEN    56065536ull
// total ~57.7 MB

__device__ __forceinline__ float leaky02(float s) {
    return s > 0.0f ? s : 0.2f * s;
}

__device__ __forceinline__ void atomicMaxF32(float* addr, float val) {
    int* ai = (int*)addr;
    int old = __float_as_int(*addr);
    while (__int_as_float(old) < val) {
        int assumed = old;
        old = atomicCAS(ai, assumed, __float_as_int(val));
        if (old == assumed) break;
    }
}

// ---------------------------------------------------------------------------
// K0: pack W (fp32 [128,256] row-major) into bf16 WMMA B lanes, same k order as A in K1.
// B tile (32x16 K x N) per (j, kc): lane l holds column n = l%16,
// k = kc*32 + (l/16)*16 + e for e = 0..15 (pairs packed per VGPR).
// Stored as Wp[((j*4+kc)*32 + lane)*16 + e] so the GEMM reads one contiguous
// 32-byte v16bf per lane (fully coalesced 1KB per wave-chunk).
// ---------------------------------------------------------------------------
__global__ void gat_pack_w(const float* __restrict__ W, __bf16* __restrict__ Wp) {
    int t = blockIdx.x * blockDim.x + threadIdx.x;      // 0 .. 32767
    if (t >= COL_TILES * K_CHUNKS * 32 * 16) return;
    int e    = t & 15;
    int lane = (t >> 4) & 31;
    int kc   = (t >> 9) & 3;
    int j    = t >> 11;
    int k    = kc * 32 + ((lane >> 4) << 4) + e;        // 0..127
    int col  = j * 16 + (lane & 15);                    // 0..255
    Wp[t] = (__bf16)W[k * GAT_F + col];
}

// ---------------------------------------------------------------------------
// K1: h = x @ W via WMMA. One wave per 16-row tile; loops 16 column tiles,
// 4 K-chunks each of v_wmma_f32_16x16x32_bf16.
// A (16x32, MxK), k order chosen here and matched by Wp: lane l row M = l%16; element e of v16bf
// holds k = (l/16)*16 + e, pairs contiguous -> float2 loads.
// D layout (32-bit C/D 16x16): lane l col N = l%16; VGPR r row M = r + (l/16)*8.
// ---------------------------------------------------------------------------
__global__ void __launch_bounds__(256)
gat_gemm_wmma(const float* __restrict__ x, const __bf16* __restrict__ Wp,
              float* __restrict__ h) {
    const int lane = threadIdx.x & 31;
    const int wave = threadIdx.x >> 5;
    const int rowTile = blockIdx.x * 8 + wave;
    if (rowTile >= ROW_TILES) return;
    const int rowBase = rowTile * 16;
    const int m = lane & 15;
    const int hiHalf = lane >> 4;                 // 0 or 1
    const float* xr = x + (size_t)(rowBase + m) * GAT_FIN;

    // Load/convert all four 16x32 A chunks for this row tile (kept in VGPRs).
    v16bf A[K_CHUNKS];
    #pragma unroll
    for (int kc = 0; kc < K_CHUNKS; ++kc) {
        #pragma unroll
        for (int v = 0; v < 8; ++v) {
            int k = kc * 32 + (hiHalf << 4) + (v << 1);
            float2 f = *(const float2*)(xr + k);
            A[kc][2 * v]     = (__bf16)f.x;
            A[kc][2 * v + 1] = (__bf16)f.y;
        }
    }

    const v16bf* wp = (const v16bf*)Wp;
    #pragma unroll 4
    for (int j = 0; j < COL_TILES; ++j) {
        v8f acc = {};
        #pragma unroll
        for (int kc = 0; kc < K_CHUNKS; ++kc) {
            v16bf b = wp[(j * K_CHUNKS + kc) * 32 + lane];
            acc = __builtin_amdgcn_wmma_f32_16x16x32_bf16(
                /*neg_a=*/false, A[kc], /*neg_b=*/false, b,
                /*c_mod=*/(short)0, acc, /*reuse_a=*/false, /*reuse_b=*/false);
        }
        const int col = j * 16 + m;
        const int mrow = hiHalf * 8;
        #pragma unroll
        for (int r = 0; r < 8; ++r) {
            h[(size_t)(rowBase + mrow + r) * GAT_F + col] = acc[r];
        }
    }
}

// ---------------------------------------------------------------------------
// K2: per-(node, head) attention scores a_src / a_dst (dot over C=32).
// ---------------------------------------------------------------------------
__global__ void gat_scores(const float* __restrict__ h,
                           const float* __restrict__ att_src,
                           const float* __restrict__ att_dst,
                           float* __restrict__ a_src, float* __restrict__ a_dst) {
    int t = blockIdx.x * blockDim.x + threadIdx.x;      // N*H
    if (t >= GAT_N * GAT_H) return;
    int n = t >> 3, hd = t & 7;
    const float* hv = h + (size_t)n * GAT_F + hd * GAT_C;
    const float* as = att_src + hd * GAT_C;
    const float* ad = att_dst + hd * GAT_C;
    float ss = 0.0f, sd = 0.0f;
    #pragma unroll
    for (int c = 0; c < GAT_C; ++c) {
        float v = hv[c];
        ss += v * as[c];
        sd += v * ad[c];
    }
    a_src[t] = ss;
    a_dst[t] = sd;
}

// K3: init segment max with the self-loop score (always present).
__global__ void gat_init_m(const float* __restrict__ a_src,
                           const float* __restrict__ a_dst,
                           float* __restrict__ m) {
    int t = blockIdx.x * blockDim.x + threadIdx.x;
    if (t >= GAT_N * GAT_H) return;
    m[t] = leaky02(a_src[t] + a_dst[t]);
}

// K4: edge-wise segment max (per edge x head).
__global__ void gat_edge_max(const int* __restrict__ ei,
                             const float* __restrict__ a_src,
                             const float* __restrict__ a_dst,
                             float* __restrict__ m) {
    int t = blockIdx.x * blockDim.x + threadIdx.x;      // E*H
    if (t >= GAT_E * GAT_H) return;
    int e = t >> 3, hd = t & 7;
    int dst = ei[e];              // edge_index[0]
    int src = ei[GAT_E + e];      // edge_index[1]
    float s = leaky02(a_src[src * GAT_H + hd] + a_dst[dst * GAT_H + hd]);
    atomicMaxF32(&m[dst * GAT_H + hd], s);
}

// K5: init denom with self-loop exp term.
__global__ void gat_init_den(const float* __restrict__ a_src,
                             const float* __restrict__ a_dst,
                             const float* __restrict__ m,
                             float* __restrict__ den) {
    int t = blockIdx.x * blockDim.x + threadIdx.x;
    if (t >= GAT_N * GAT_H) return;
    float s = leaky02(a_src[t] + a_dst[t]);
    den[t] = __expf(s - m[t]);
}

// K6: accumulate edge exp terms into denom.
__global__ void gat_edge_den(const int* __restrict__ ei,
                             const float* __restrict__ a_src,
                             const float* __restrict__ a_dst,
                             const float* __restrict__ m,
                             float* __restrict__ den) {
    int t = blockIdx.x * blockDim.x + threadIdx.x;      // E*H
    if (t >= GAT_E * GAT_H) return;
    int e = t >> 3, hd = t & 7;
    int dst = ei[e];
    int src = ei[GAT_E + e];
    int seg = dst * GAT_H + hd;
    float s = leaky02(a_src[src * GAT_H + hd] + a_dst[seg]);
    atomicAdd(&den[seg], __expf(s - m[seg]));
}

// K7: out = self-loop contribution + bias (initializes d_out fully).
__global__ void gat_out_self(const float* __restrict__ h,
                             const float* __restrict__ a_src,
                             const float* __restrict__ a_dst,
                             const float* __restrict__ m,
                             const float* __restrict__ den,
                             const float* __restrict__ dp_self,
                             const float* __restrict__ bias,
                             float* __restrict__ out) {
    int t = blockIdx.x * blockDim.x + threadIdx.x;      // N*256
    if (t >= GAT_N * GAT_F) return;
    int n = t >> 8, f = t & 255, hd = f >> 5;
    int seg = n * GAT_H + hd;
    float s = leaky02(a_src[seg] + a_dst[seg]);
    float p = __expf(s - m[seg]) / den[seg] * dp_self[seg];
    out[t] = p * h[t] + bias[f];
}

// K8: edge aggregation. One wave per edge: lane = channel c (C==32==wave32),
// loop over 8 heads; h gather + out atomics stay resident in the 192MB L2.
__global__ void __launch_bounds__(256)
gat_edge_agg(const int* __restrict__ ei,
             const float* __restrict__ h,
             const float* __restrict__ a_src,
             const float* __restrict__ a_dst,
             const float* __restrict__ m,
             const float* __restrict__ den,
             const float* __restrict__ dp,
             float* __restrict__ out) {
    int e = blockIdx.x * 8 + (threadIdx.x >> 5);
    int lane = threadIdx.x & 31;
    if (e >= GAT_E) return;
    int dst = ei[e];
    int src = ei[GAT_E + e];
    const float* hs = h + (size_t)src * GAT_F;
    float* od = out + (size_t)dst * GAT_F;
    #pragma unroll
    for (int hd = 0; hd < GAT_H; ++hd) {
        int seg = dst * GAT_H + hd;
        float s = leaky02(a_src[src * GAT_H + hd] + a_dst[seg]);
        float p = __expf(s - m[seg]) / den[seg] * dp[e * GAT_H + hd];
        int f = hd * GAT_C + lane;
        atomicAdd(&od[f], p * hs[f]);
    }
}

// ---------------------------------------------------------------------------
extern "C" void kernel_launch(void* const* d_in, const int* in_sizes, int n_in,
                              void* d_out, int out_size, void* d_ws, size_t ws_size,
                              hipStream_t stream) {
    const float* x        = (const float*)d_in[0];     // [N,128]
    const int*   ei       = (const int*)  d_in[1];     // [2,E]
    const float* dp       = (const float*)d_in[2];     // [E,H]
    const float* dp_self  = (const float*)d_in[3];     // [N,H]
    const float* W        = (const float*)d_in[4];     // [128,256]
    const float* att_src  = (const float*)d_in[5];     // [H,C]
    const float* att_dst  = (const float*)d_in[6];     // [H,C]
    const float* bias     = (const float*)d_in[7];     // [256]
    float* out = (float*)d_out;                        // [N,256]

    char* ws = (char*)d_ws;
    float*  h     = (float*)(ws + OFF_H);
    __bf16* Wp    = (__bf16*)(ws + OFF_WP);
    float*  a_src = (float*)(ws + OFF_ASRC);
    float*  a_dst = (float*)(ws + OFF_ADST);
    float*  mbuf  = (float*)(ws + OFF_M);
    float*  den   = (float*)(ws + OFF_DEN);

    const int NH = GAT_N * GAT_H;       // 400,000
    const int EH = GAT_E * GAT_H;       // 6,400,000
    const int NF = GAT_N * GAT_F;       // 12,800,000

    gat_pack_w   <<<(COL_TILES * K_CHUNKS * 32 * 16 + 255) / 256, 256, 0, stream>>>(W, Wp);
    gat_gemm_wmma<<<(ROW_TILES + 7) / 8, 256, 0, stream>>>(x, Wp, h);
    gat_scores   <<<(NH + 255) / 256, 256, 0, stream>>>(h, att_src, att_dst, a_src, a_dst);
    gat_init_m   <<<(NH + 255) / 256, 256, 0, stream>>>(a_src, a_dst, mbuf);
    gat_edge_max <<<(EH + 255) / 256, 256, 0, stream>>>(ei, a_src, a_dst, mbuf);
    gat_init_den <<<(NH + 255) / 256, 256, 0, stream>>>(a_src, a_dst, mbuf, den);
    gat_edge_den <<<(EH + 255) / 256, 256, 0, stream>>>(ei, a_src, a_dst, mbuf, den);
    gat_out_self <<<(NF + 255) / 256, 256, 0, stream>>>(h, a_src, a_dst, mbuf, den,
                                                        dp_self, bias, out);
    gat_edge_agg <<<(GAT_E + 7) / 8, 256, 0, stream>>>(ei, h, a_src, a_dst, mbuf, den,
                                                       dp, out);
    (void)in_sizes; (void)n_in; (void)out_size; (void)ws_size;
}
